// PathWeightModule_49778670961190
// MI455X (gfx1250) — matrix operand order repaired and checked by reading, not compiled, not run on hardware
//
#include <hip/hip_runtime.h>
#include <hip/hip_bf16.h>

typedef __attribute__((ext_vector_type(16))) _Float16 v16h;
typedef __attribute__((ext_vector_type(8)))  float    v8f;

#define DD   128
#define HC   192
#define CC   64
#define HH   3
#define NEG_SLOPE 0.2f

// ---------------------------------------------------------------------------
// Zero-fill scratch (lmaxU / denom / outacc region)
// ---------------------------------------------------------------------------
__global__ void zero_words(unsigned* __restrict__ p, long n) {
    long i = (long)blockIdx.x * blockDim.x + threadIdx.x;
    if (i < n) p[i] = 0u;
}

// ---------------------------------------------------------------------------
// WMMA GEMM:  Y[nRows x 192] = X[nRows x 128] * W[128 x 192] + bias
// One wave computes a 16x64 strip: 4 accumulators share one A fragment,
// k-loop fully unrolled -> 16 static v_wmma ops, A read once per 4 WMMAs.
// ---------------------------------------------------------------------------
__global__ void gemm_wmma_f16(const float* __restrict__ X,
                              const float* __restrict__ W,
                              const float* __restrict__ bias,
                              float* __restrict__ Y,
                              int nRows) {
    const int wid  = (int)((blockIdx.x * blockDim.x + threadIdx.x) >> 5);
    const int lane = threadIdx.x & 31;
    const int colGroups = HC / 64;                    // 3
    const int rowTiles  = (nRows + 15) / 16;
    if (wid >= rowTiles * colGroups) return;

    const int rowBase = (wid / colGroups) << 4;
    const int colBase = (wid % colGroups) << 6;       // 64-wide strip
    const int hi      = lane >> 4;                    // 0: lanes 0-15, 1: lanes 16-31
    const int lane16  = lane & 15;

    // Clamp A row instead of predicating loads: duplicate rows are loaded
    // but never stored, so the math stays correct and loads stay uniform.
    int arow = rowBase + lane16;
    if (arow >= nRows) arow = nRows - 1;
    const float4* __restrict__ apv =
        (const float4*)(X + (size_t)arow * DD);       // DD % 4 == 0

    v8f acc0 = {}, acc1 = {}, acc2 = {}, acc3 = {};
#pragma unroll
    for (int k0 = 0; k0 < DD; k0 += 32) {
        // ---- A tile (16x32 f16), lane K-base = k0 + 8*hi (multiple of 8)
        const int kb4 = (k0 + (hi ? 8 : 0)) >> 2;     // float4 index
        const float4 a0 = apv[kb4 + 0];               // K = kb .. kb+3
        const float4 a1 = apv[kb4 + 1];               // K = kb+4 .. kb+7
        const float4 a2 = apv[kb4 + 4];               // K = kb+16 .. kb+19
        const float4 a3 = apv[kb4 + 5];               // K = kb+20 .. kb+23
        v16h a;
        a[0]  = (_Float16)a0.x;  a[1]  = (_Float16)a0.y;
        a[2]  = (_Float16)a0.z;  a[3]  = (_Float16)a0.w;
        a[4]  = (_Float16)a1.x;  a[5]  = (_Float16)a1.y;
        a[6]  = (_Float16)a1.z;  a[7]  = (_Float16)a1.w;
        a[8]  = (_Float16)a2.x;  a[9]  = (_Float16)a2.y;
        a[10] = (_Float16)a2.z;  a[11] = (_Float16)a2.w;
        a[12] = (_Float16)a3.x;  a[13] = (_Float16)a3.y;
        a[14] = (_Float16)a3.z;  a[15] = (_Float16)a3.w;

        // ---- 4 B tiles (32x16 f16 each): col N=lane16, lane K-base
        //      kb = k0 + 8*hi, K = kb..kb+7, kb+16..kb+23 (strided rows of W)
        const float* bbase =
            W + (size_t)(k0 + (hi ? 8 : 0)) * HC + colBase + lane16;
#pragma unroll
        for (int t = 0; t < 4; ++t) {
            const float* bp = bbase + t * 16;
            v16h b;
#pragma unroll
            for (int i = 0; i < 16; ++i)
                b[i] = (_Float16)bp[(size_t)(i + (i >= 8 ? 8 : 0)) * HC];
            v8f& acc = (t == 0) ? acc0 : (t == 1) ? acc1 : (t == 2) ? acc2 : acc3;
            acc = __builtin_amdgcn_wmma_f32_16x16x32_f16(
                      false, a, false, b, (short)0, acc, false, false);
        }
    }

    // ---- C/D store: VGPR j -> M = j + 8*hi, N = lane16 (per 16-col tile)
#pragma unroll
    for (int t = 0; t < 4; ++t) {
        const v8f& acc = (t == 0) ? acc0 : (t == 1) ? acc1 : (t == 2) ? acc2 : acc3;
        const int n  = colBase + t * 16 + lane16;
        const float bv = bias[n];
#pragma unroll
        for (int j = 0; j < 8; ++j) {
            const int m = rowBase + j + (hi ? 8 : 0);
            if (m < nRows) Y[(size_t)m * HC + n] = acc[j] + bv;
        }
    }
}

// ---------------------------------------------------------------------------
// Order-preserving float <-> uint for atomicMax-based segment max
// ---------------------------------------------------------------------------
__device__ __forceinline__ unsigned f2ord(float f) {
    unsigned u = __float_as_uint(f);
    return (u & 0x80000000u) ? ~u : (u | 0x80000000u);
}
__device__ __forceinline__ float ord2f(unsigned u) {
    return __uint_as_float((u & 0x80000000u) ? (u ^ 0x80000000u) : ~u);
}

// ---------------------------------------------------------------------------
// Pass 1 over edges: per (e,h) attention logit + segment max (atomicMax)
// ---------------------------------------------------------------------------
__global__ void edge_logits(const int*   __restrict__ ei,
                            const float* __restrict__ eattr,
                            const float* __restrict__ sim_w,
                            const float* __restrict__ W_e,
                            const float* __restrict__ att,
                            const float* __restrict__ XL,
                            const float* __restrict__ XR,
                            float*    __restrict__ logits,
                            unsigned* __restrict__ lmaxU,
                            int E) {
    int idx = blockIdx.x * blockDim.x + threadIdx.x;
    if (idx >= E * HH) return;
    const int e = idx / HH;
    const int h = idx - e * HH;
    const int src = ei[e];
    const int dst = ei[E + e];

    // softmax over the 3 similarity channels (uniform scalar work)
    const float s0 = sim_w[0], s1 = sim_w[1], s2 = sim_w[2];
    const float mx = fmaxf(s0, fmaxf(s1, s2));
    const float e0 = __expf(s0 - mx), e1 = __expf(s1 - mx), e2 = __expf(s2 - mx);
    const float inv = 1.0f / (e0 + e1 + e2);
    const float fused = (eattr[e * 3 + 0] * e0 +
                         eattr[e * 3 + 1] * e1 +
                         eattr[e * 3 + 2] * e2) * inv;

    const float* xl = XL  + (size_t)src * HC + h * CC;
    const float* xr = XR  + (size_t)dst * HC + h * CC;
    const float* we = W_e + h * CC;
    const float* at = att + h * CC;

    float logit = 0.0f;
#pragma unroll 8
    for (int c = 0; c < CC; ++c) {
        float m = xl[c] + xr[c] + fused * we[c];
        m = (m > 0.0f) ? m : NEG_SLOPE * m;
        logit += m * at[c];
    }
    logits[idx] = logit;
    atomicMax(&lmaxU[dst * HH + h], f2ord(logit));
}

// ---------------------------------------------------------------------------
// Pass 2: ex = exp(logit - lmax[dst]); denom[dst] += ex  (in-place overwrite)
// ---------------------------------------------------------------------------
__global__ void edge_exp(const int*      __restrict__ ei,
                         float*          __restrict__ logits_ex,
                         const unsigned* __restrict__ lmaxU,
                         float*          __restrict__ denom,
                         int E) {
    int idx = blockIdx.x * blockDim.x + threadIdx.x;
    if (idx >= E * HH) return;
    const int e = idx / HH;
    const int h = idx - e * HH;
    const int dst = ei[E + e];
    const float lm = ord2f(lmaxU[dst * HH + h]);
    const float ex = __expf(logits_ex[idx] - lm);
    logits_ex[idx] = ex;
    atomicAdd(&denom[dst * HH + h], ex);
}

// ---------------------------------------------------------------------------
// Pass 3: outacc[dst,c] += (1/H) * sum_h alpha(e,h) * XL[src, h*C + c]
// ---------------------------------------------------------------------------
__global__ void edge_aggregate(const int*   __restrict__ ei,
                               const float* __restrict__ ex,
                               const float* __restrict__ denom,
                               const float* __restrict__ XL,
                               float*       __restrict__ outacc,
                               int E) {
    int idx = blockIdx.x * blockDim.x + threadIdx.x;
    if (idx >= E * CC) return;
    const int e = idx >> 6;
    const int c = idx & (CC - 1);
    const int src = ei[e];
    const int dst = ei[E + e];

    float acc = 0.0f;
#pragma unroll
    for (int h = 0; h < HH; ++h) {
        const float d     = denom[dst * HH + h];
        const float alpha = ex[e * HH + h] / fmaxf(d, 1e-16f);
        acc += alpha * XL[(size_t)src * HC + h * CC + c];
    }
    atomicAdd(&outacc[(size_t)dst * CC + c], acc * (1.0f / (float)HH));
}

// ---------------------------------------------------------------------------
// Per-node linear: scores[n] = lin_b + sum_c (outacc[n,c] + gat_b[c]) * lin_w[c]
// ---------------------------------------------------------------------------
__global__ void node_scores(const float* __restrict__ outacc,
                            const float* __restrict__ gat_b,
                            const float* __restrict__ lin_w,
                            const float* __restrict__ lin_b,
                            float* __restrict__ scores,
                            int N) {
    int n = blockIdx.x * blockDim.x + threadIdx.x;
    if (n >= N) return;
    float s = lin_b[0];
#pragma unroll 8
    for (int c = 0; c < CC; ++c)
        s += (outacc[(size_t)n * CC + c] + gat_b[c]) * lin_w[c];
    scores[n] = s;
}

// ---------------------------------------------------------------------------
// Global softmax over N scores, scaled by k_weight. Single block, LDS tree.
// ---------------------------------------------------------------------------
__global__ void final_softmax(const float* __restrict__ scores,
                              const float* __restrict__ k_weight,
                              float* __restrict__ out,
                              int N) {
    __shared__ float red[1024];
    const int t = threadIdx.x;
    const int B = blockDim.x;

    float mx = -INFINITY;
    for (int i = t; i < N; i += B) mx = fmaxf(mx, scores[i]);
    red[t] = mx; __syncthreads();
    for (int s = B >> 1; s > 0; s >>= 1) {
        if (t < s) red[t] = fmaxf(red[t], red[t + s]);
        __syncthreads();
    }
    mx = red[0]; __syncthreads();

    float sum = 0.0f;
    for (int i = t; i < N; i += B) sum += __expf(scores[i] - mx);
    red[t] = sum; __syncthreads();
    for (int s = B >> 1; s > 0; s >>= 1) {
        if (t < s) red[t] += red[t + s];
        __syncthreads();
    }
    const float inv = 1.0f / red[0];

    for (int i = t; i < N; i += B)
        out[i] = k_weight[i] * __expf(scores[i] - mx) * inv;
}

// ---------------------------------------------------------------------------
// Launch
// ---------------------------------------------------------------------------
extern "C" void kernel_launch(void* const* d_in, const int* in_sizes, int n_in,
                              void* d_out, int out_size, void* d_ws, size_t ws_size,
                              hipStream_t stream) {
    const int*   edge_index = (const int*)  d_in[0];
    const float* edge_attr  = (const float*)d_in[1];
    const float* sim_w      = (const float*)d_in[2];
    const float* x_emb      = (const float*)d_in[3];
    const float* W_l        = (const float*)d_in[4];
    const float* b_l        = (const float*)d_in[5];
    const float* W_r        = (const float*)d_in[6];
    const float* b_r        = (const float*)d_in[7];
    const float* W_e        = (const float*)d_in[8];
    const float* att        = (const float*)d_in[9];
    const float* gat_b      = (const float*)d_in[10];
    const float* lin_w      = (const float*)d_in[11];
    const float* lin_b      = (const float*)d_in[12];
    const float* k_weight   = (const float*)d_in[13];

    const int E = in_sizes[0] / 2;
    const int N = in_sizes[3] / DD;

    // ---- workspace layout (floats); lmaxU/denom/outacc contiguous -------
    float* ws       = (float*)d_ws;
    float* XL       = ws;                                   // N*192
    float* XR       = XL + (size_t)N * HC;                  // N*192
    float* exbuf    = XR + (size_t)N * HC;                  // E*3 (logits, then ex)
    unsigned* lmaxU = (unsigned*)(exbuf + (size_t)E * HH);  // N*3
    float* denom    = (float*)(lmaxU + (size_t)N * HH);     // N*3
    float* outacc   = denom + (size_t)N * HH;               // N*64
    float* scores   = outacc + (size_t)N * CC;              // N

    // ---- zero the atomic-accumulated region -----------------------------
    const long zn = (long)N * HH + (long)N * HH + (long)N * CC;
    zero_words<<<(int)((zn + 255) / 256), 256, 0, stream>>>(lmaxU, zn);

    // ---- dense GEMMs via WMMA -------------------------------------------
    const int rowTiles  = (N + 15) / 16;
    const int colGroups = HC / 64;                    // 3
    const int waves     = rowTiles * colGroups;
    const int gBlocks   = (waves * 32 + 255) / 256;
    gemm_wmma_f16<<<gBlocks, 256, 0, stream>>>(x_emb, W_l, b_l, XL, N);
    gemm_wmma_f16<<<gBlocks, 256, 0, stream>>>(x_emb, W_r, b_r, XR, N);

    // ---- edge phase ------------------------------------------------------
    const int ehThreads = E * HH;
    const int ehBlocks  = (ehThreads + 255) / 256;
    edge_logits<<<ehBlocks, 256, 0, stream>>>(edge_index, edge_attr, sim_w, W_e, att,
                                              XL, XR, exbuf, lmaxU, E);
    edge_exp<<<ehBlocks, 256, 0, stream>>>(edge_index, exbuf, lmaxU, denom, E);

    const int aggThreads = E * CC;
    const int aggBlocks  = (aggThreads + 255) / 256;
    edge_aggregate<<<aggBlocks, 256, 0, stream>>>(edge_index, exbuf, denom, XL,
                                                  outacc, E);

    // ---- node phase ------------------------------------------------------
    node_scores<<<(N + 255) / 256, 256, 0, stream>>>(outacc, gat_b, lin_w, lin_b,
                                                     scores, N);
    final_softmax<<<1, 1024, 0, stream>>>(scores, k_weight, (float*)d_out, N);
}
